// SwinTransformerBlock3D_34205119545826
// MI455X (gfx1250) — matrix operand order repaired and checked by reading, not compiled, not run on hardware
//
#include <hip/hip_runtime.h>
#include <hip/hip_fp16.h>

typedef _Float16 half_t;
typedef __attribute__((ext_vector_type(16))) _Float16 v16h;
typedef __attribute__((ext_vector_type(8)))  _Float16 v8h;
typedef __attribute__((ext_vector_type(8)))  float    v8f;

#define DIM      192
#define HEADS    6
#define HEAD_DIM 32
#define NTOK     64
#define NWIN     4096
#define SHIFTV   2
#define QSCALE   0.17677669529663687f   // 32^-0.5
#define LN_EPS   1e-5f

// ---- workspace layout (bytes) ----
#define OFF_QKVW 0u                         // 576*192 f16 = 221184
#define OFF_PROJW 221184u                   // 192*192 f16 = 73728
#define OFF_FC1W  294912u                   // 768*192 f16 = 294912
#define OFF_FC2W  589824u                   // 192*768 f16 = 294912
#define OFF_BIAS  884736u                   // 6*64*64 f32 = 98304
#define OFF_XW    983040u                   // 4096*64*192 f16 = 100663296

// ================= WMMA helpers (gfx1250, wave32) =================
__device__ __forceinline__ v8f wmma_f16(v16h a, v16h b, v8f c) {
    return __builtin_amdgcn_wmma_f32_16x16x32_f16(
        /*neg_a=*/false, a, /*neg_b=*/false, b,
        /*c_mod=*/(short)0, c, /*reuse_a=*/false, /*reuse_b=*/false);
}

__device__ __forceinline__ v16h cat8(v8h lo, v8h hi) {
    return __builtin_shufflevector(lo, hi, 0,1,2,3,4,5,6,7,8,9,10,11,12,13,14,15);
}

// A-fragment (16x32 f16): lane L holds row r0+(L&15);
// elem j = K k0 + ((L>>4)*16) + j (contiguous).
// Requires 16B alignment of S + row*ld + k0 + 16*(L>>4)  (ld, k0 multiples of 8).
__device__ __forceinline__ v16h frag_a(const half_t* S, int ld, int r0, int k0, int lane) {
    const half_t* p = S + (r0 + (lane & 15)) * ld + k0 + ((lane >> 4) << 4);
    v8h lo = *(const v8h*)p;
    v8h hi = *(const v8h*)(p + 8);
    return cat8(lo, hi);
}

// B-fragment (32x16 f16), contiguous-K source: B[k][n] = base[n*sn + k].
// lane L holds col n0+(L&15); elem j = B[k0 + ((L>>4)*16) + j][n].
__device__ __forceinline__ v16h frag_bc(const half_t* base, long sn, int k0, int n0, int lane) {
    const half_t* p = base + (long)(n0 + (lane & 15)) * sn + (k0 + ((lane >> 4) << 4));
    v8h lo = *(const v8h*)p;
    v8h hi = *(const v8h*)(p + 8);
    return cat8(lo, hi);
}

// C/D fragment element r, lane L -> (M = r + 8*(L>>4), N = L&15)

// ================= prep kernels =================
__global__ void cvt_f16_kernel(const float* __restrict__ src, half_t* __restrict__ dst, int n) {
    int i = blockIdx.x * 256 + threadIdx.x;
    if (i < n) dst[i] = (half_t)src[i];
}

__global__ void build_bias_kernel(const float* __restrict__ rpb, float* __restrict__ bias) {
    int i = blockIdx.x * 256 + threadIdx.x;      // 4096 = 64*64 (n,m) pairs
    if (i >= 64 * 64) return;
    int n = i >> 6, m = i & 63;
    int n0 = n >> 4, n1 = (n >> 2) & 3, n2 = n & 3;
    int m0 = m >> 4, m1 = (m >> 2) & 3, m2 = m & 3;
    int idx = (n0 - m0 + 3) * 49 + (n1 - m1 + 3) * 7 + (n2 - m2 + 3);
#pragma unroll
    for (int h = 0; h < HEADS; ++h)
        bias[h * 4096 + n * 64 + m] = rpb[idx * HEADS + h];
}

// ================= LN1 + roll(-shift) + window partition -> f16 =================
__global__ void __launch_bounds__(256) ln1_window_kernel(
    const float* __restrict__ x, const float* __restrict__ g, const float* __restrict__ b,
    half_t* __restrict__ xw) {
    int wid = threadIdx.x >> 5, lane = threadIdx.x & 31;
    int tok = blockIdx.x * 8 + wid;              // 0..262143
    int w = tok >> 6, t = tok & 63;
    int wdb = w >> 8, whb = (w >> 4) & 15, wwb = w & 15;
    int td = t >> 4, th = (t >> 2) & 3, tw = t & 3;
    int sd = ((wdb << 2) + td + SHIFTV) & 63;    // roll(-shift): src = pos + shift
    int sh = ((whb << 2) + th + SHIFTV) & 63;
    int sw = ((wwb << 2) + tw + SHIFTV) & 63;
    const float* row = x + ((long)((sd * 64 + sh) * 64 + sw)) * DIM;
    float v[6], s = 0.f, ss = 0.f;
#pragma unroll
    for (int j = 0; j < 6; ++j) {
        float f = row[lane + 32 * j];
        v[j] = f; s += f; ss += f * f;
    }
#pragma unroll
    for (int off = 16; off > 0; off >>= 1) {
        s  += __shfl_xor(s,  off, 32);
        ss += __shfl_xor(ss, off, 32);
    }
    float mean = s * (1.f / DIM);
    float var  = ss * (1.f / DIM) - mean * mean;
    float inv  = rsqrtf(var + LN_EPS);
    half_t* outp = xw + (long)tok * DIM;
#pragma unroll
    for (int j = 0; j < 6; ++j) {
        int c = lane + 32 * j;
        outp[c] = (half_t)((v[j] - mean) * inv * g[c] + b[c]);
    }
}

// ================= fused attention: QKV -> scores -> softmax -> PV -> proj -> residual =================
__global__ void __launch_bounds__(256) attn_kernel(
    const float* __restrict__ x, const float* __restrict__ mask,
    const half_t* __restrict__ xw_h,
    const half_t* __restrict__ qkvw, const float* __restrict__ qkvb,
    const half_t* __restrict__ projw, const float* __restrict__ projb,
    const float* __restrict__ bias, float* __restrict__ out) {
    __shared__ half_t sx[64 * 208];                     // 26.6 KB
    __shared__ half_t sqkv[64 * 40 + 64 * 40 + 32 * 72]; // q,k [tok][d]; vT [d][tok]; 14.8 KB
    __shared__ float  ssc[64 * 68];                     // scores f32 (O_h reuses as f16), 17.4 KB

    const int w = blockIdx.x;
    const int tid = threadIdx.x, wid = tid >> 5, lane = tid & 31;

    // vectorized tile fill: 1536 v8h chunks, 8-half chunks never cross a row
    const half_t* src = xw_h + (long)w * NTOK * DIM;
    for (int i = tid; i < (NTOK * DIM) / 8; i += 256) {
        int idx = i * 8;
        int r = idx / DIM, c = idx - r * DIM;
        *(v8h*)(sx + r * 208 + c) = *(const v8h*)(src + idx);
    }
    __syncthreads();

    v8f oacc[6];
#pragma unroll
    for (int i = 0; i < 6; ++i) oacc[i] = (v8f){0.f,0.f,0.f,0.f,0.f,0.f,0.f,0.f};

    half_t* pq  = sqkv;                 // 64 x 40
    half_t* pk  = sqkv + 64 * 40;       // 64 x 40
    half_t* pvT = sqkv + 2 * 64 * 40;   // 32 x 72 (transposed V: [dim][token])
    half_t* pp  = sqkv;                 // probs f16 64x72, reuses q+k region
    half_t* po  = (half_t*)ssc;         // O_h f16 64x40, reuses score region

    for (int h = 0; h < HEADS; ++h) {
        // ---- QKV for this head: 24 tiles (q/k/v each 4x2), 3 per wave, K-loop 6 ----
#pragma unroll
        for (int i = 0; i < 3; ++i) {
            int t  = wid * 3 + i;
            int mat = t >> 3;                 // 0=q 1=k 2=v
            int tt = t & 7, tr = tt >> 1, tc = tt & 1;
            const half_t* wb = qkvw + (long)(mat * DIM + h * HEAD_DIM) * DIM;
            v8f acc = (v8f){0.f,0.f,0.f,0.f,0.f,0.f,0.f,0.f};
#pragma unroll
            for (int kk = 0; kk < 6; ++kk) {
                v16h a = frag_a(sx, 208, tr * 16, kk * 32, lane);
                v16h b = frag_bc(wb, DIM, kk * 32, tc * 16, lane); // B[k][n]=W[n][k]
                acc = wmma_f16(a, b, acc);
            }
            int n = tc * 16 + (lane & 15);
            float bia = qkvb[mat * DIM + h * HEAD_DIM + n];
            int mrow = tr * 16 + ((lane >> 4) << 3);
            if (mat == 2) {
                // store V transposed: vT[n][m..m+7] contiguous -> one b128 store
                v8h pv8;
#pragma unroll
                for (int r = 0; r < 8; ++r) pv8[r] = (half_t)(acc[r] + bia);
                *(v8h*)(pvT + n * 72 + mrow) = pv8;
            } else {
                half_t* dst = (mat == 0) ? pq : pk;
                float sc = (mat == 0) ? QSCALE : 1.f;
#pragma unroll
                for (int r = 0; r < 8; ++r)
                    dst[(mrow + r) * 40 + n] = (half_t)((acc[r] + bia) * sc);
            }
        }
        __syncthreads();

        // ---- scores S = q @ k^T : 16 tiles, 2 per wave, K = 32 (one WMMA) ----
#pragma unroll
        for (int i = 0; i < 2; ++i) {
            int t = wid * 2 + i, tr = t >> 2, tc = t & 3;
            v16h a = frag_a(pq, 40, tr * 16, 0, lane);
            v16h b = frag_bc(pk, 40, 0, tc * 16, lane);  // B[k][n]=K[n][k], contiguous
            v8f acc = (v8f){0.f,0.f,0.f,0.f,0.f,0.f,0.f,0.f};
            acc = wmma_f16(a, b, acc);
            int n = tc * 16 + (lane & 15);
            int mrow = tr * 16 + ((lane >> 4) << 3);
#pragma unroll
            for (int r = 0; r < 8; ++r) ssc[(mrow + r) * 68 + n] = acc[r];
        }
        __syncthreads();

        // ---- softmax: 4 lanes per row (all 256 threads), bias + mask fused ----
        {
            int m = tid >> 2, part = (tid & 3) * 16;
            const float* bi = bias + h * 4096 + m * 64 + part;
            const float* ma = mask + (long)w * 4096 + m * 64 + part;
            float* row = ssc + m * 68 + part;
            float v[16];
            float mx = -1e30f;
#pragma unroll
            for (int j = 0; j < 16; ++j) {
                v[j] = row[j] + bi[j] + ma[j];
                mx = fmaxf(mx, v[j]);
            }
            mx = fmaxf(mx, __shfl_xor(mx, 1, 4));
            mx = fmaxf(mx, __shfl_xor(mx, 2, 4));
            float sum = 0.f;
#pragma unroll
            for (int j = 0; j < 16; ++j) { v[j] = __expf(v[j] - mx); sum += v[j]; }
            sum += __shfl_xor(sum, 1, 4);
            sum += __shfl_xor(sum, 2, 4);
            float inv = 1.f / sum;
            v8h p0, p1;
#pragma unroll
            for (int j = 0; j < 8; ++j) {
                p0[j] = (half_t)(v[j] * inv);
                p1[j] = (half_t)(v[j + 8] * inv);
            }
            __syncthreads();   // q/k reads done before pp overwrites that region
            *(v8h*)(pp + m * 72 + part)     = p0;
            *(v8h*)(pp + m * 72 + part + 8) = p1;
        }
        __syncthreads();

        // ---- O_h = P @ V : 8 tiles, 1 per wave, K-loop 2 ----
        {
            int tr = wid >> 1, tc = wid & 1;
            v8f acc = (v8f){0.f,0.f,0.f,0.f,0.f,0.f,0.f,0.f};
#pragma unroll
            for (int kk = 0; kk < 2; ++kk) {
                v16h a = frag_a(pp, 72, tr * 16, kk * 32, lane);
                v16h b = frag_bc(pvT, 72, kk * 32, tc * 16, lane); // B[k][n]=vT[n][k]
                acc = wmma_f16(a, b, acc);
            }
            int n = tc * 16 + (lane & 15);
            int mrow = tr * 16 + ((lane >> 4) << 3);
#pragma unroll
            for (int r = 0; r < 8; ++r) po[(mrow + r) * 40 + n] = (half_t)acc[r];
        }
        __syncthreads();

        // ---- out(64x192) += O_h @ projW[:, h-slice]^T : 48 tiles, 6 per wave ----
#pragma unroll
        for (int i = 0; i < 6; ++i) {
            int t = wid * 6 + i, tr = t / 12, tc = t % 12;
            v16h a = frag_a(po, 40, tr * 16, 0, lane);
            // B[k][n] = projW[n][h*32+k], contiguous in k
            v16h b = frag_bc(projw + h * HEAD_DIM, DIM, 0, tc * 16, lane);
            oacc[i] = wmma_f16(a, b, oacc[i]);
        }
        __syncthreads();
    }

    // ---- window_reverse + roll(+shift) + residual, write x1 into d_out ----
    int wdb = w >> 8, whb = (w >> 4) & 15, wwb = w & 15;
#pragma unroll
    for (int i = 0; i < 6; ++i) {
        int t = wid * 6 + i, tr = t / 12, tc = t % 12;
        int n = tc * 16 + (lane & 15);
        int mbase = tr * 16 + ((lane >> 4) << 3);
        float pb = projb[n];
#pragma unroll
        for (int r = 0; r < 8; ++r) {
            int m = mbase + r;
            int td = m >> 4, th = (m >> 2) & 3, tw = m & 3;
            int pd = ((wdb << 2) + td + SHIFTV) & 63;
            int ph = ((whb << 2) + th + SHIFTV) & 63;
            int pw = ((wwb << 2) + tw + SHIFTV) & 63;
            long lin = ((long)((pd * 64 + ph) * 64 + pw)) * DIM + n;
            out[lin] = x[lin] + oacc[i][r] + pb;
        }
    }
}

// ================= LN2 + MLP (fc1 -> GELU -> fc2) + residual (in-place on d_out) =================
__global__ void __launch_bounds__(256) mlp_kernel(
    float* __restrict__ xo,
    const float* __restrict__ g, const float* __restrict__ b,
    const half_t* __restrict__ w1, const float* __restrict__ b1,
    const half_t* __restrict__ w2, const float* __restrict__ b2) {
    __shared__ half_t sxh[64 * 208];   // 26.6 KB
    __shared__ half_t shh[64 * 72];    // 9.2 KB

    long r0 = (long)blockIdx.x * 64;
    int tid = threadIdx.x, wid = tid >> 5, lane = tid & 31;

    // LN2: one wave per row, 8 rounds
#pragma unroll 1
    for (int rr = 0; rr < 8; ++rr) {
        int row = rr * 8 + wid;
        const float* rp = xo + (r0 + row) * DIM;
        float v[6], s = 0.f, ss = 0.f;
#pragma unroll
        for (int j = 0; j < 6; ++j) {
            float f = rp[lane + 32 * j];
            v[j] = f; s += f; ss += f * f;
        }
#pragma unroll
        for (int off = 16; off > 0; off >>= 1) {
            s  += __shfl_xor(s,  off, 32);
            ss += __shfl_xor(ss, off, 32);
        }
        float mean = s * (1.f / DIM);
        float var  = ss * (1.f / DIM) - mean * mean;
        float inv  = rsqrtf(var + LN_EPS);
#pragma unroll
        for (int j = 0; j < 6; ++j) {
            int c = lane + 32 * j;
            sxh[row * 208 + c] = (half_t)((v[j] - mean) * inv * g[c] + b[c]);
        }
    }
    __syncthreads();

    v8f yacc[6];
#pragma unroll
    for (int i = 0; i < 6; ++i) yacc[i] = (v8f){0.f,0.f,0.f,0.f,0.f,0.f,0.f,0.f};

#pragma unroll 1
    for (int ch = 0; ch < 12; ++ch) {          // 768 = 12 chunks of 64
        // hchunk = gelu(sxh @ w1[ch]^T + b1) : 16 tiles, 2 per wave, K-loop 6
#pragma unroll
        for (int i = 0; i < 2; ++i) {
            int t = wid * 2 + i, tr = t >> 2, tc = t & 3;
            const half_t* wb = w1 + (long)(ch * 64) * DIM;    // B[k][n]=W1[ch*64+n][k]
            v8f acc = (v8f){0.f,0.f,0.f,0.f,0.f,0.f,0.f,0.f};
#pragma unroll
            for (int kk = 0; kk < 6; ++kk) {
                v16h a = frag_a(sxh, 208, tr * 16, kk * 32, lane);
                v16h bb = frag_bc(wb, DIM, kk * 32, tc * 16, lane);
                acc = wmma_f16(a, bb, acc);
            }
            int n = tc * 16 + (lane & 15);
            float bia = b1[ch * 64 + n];
            int mrow = tr * 16 + ((lane >> 4) << 3);
#pragma unroll
            for (int r = 0; r < 8; ++r) {
                float v = acc[r] + bia;
                v = 0.5f * v * (1.f + erff(v * 0.70710678118654752f)); // exact GELU
                shh[(mrow + r) * 72 + n] = (half_t)v;
            }
        }
        __syncthreads();

        // yacc += hchunk @ w2^T-chunk : 48 tiles, 6 per wave, K-loop 2
#pragma unroll
        for (int i = 0; i < 6; ++i) {
            int t = wid * 6 + i, tr = t / 12, tc = t % 12;
            const half_t* wb = w2 + ch * 64;   // B[k][n]=W2[n][ch*64+k], contiguous
#pragma unroll
            for (int kk = 0; kk < 2; ++kk) {
                v16h a = frag_a(shh, 72, tr * 16, kk * 32, lane);
                v16h bb = frag_bc(wb, 768, kk * 32, tc * 16, lane);
                yacc[i] = wmma_f16(a, bb, yacc[i]);
            }
        }
        __syncthreads();
    }

    // final residual add in-place
#pragma unroll
    for (int i = 0; i < 6; ++i) {
        int t = wid * 6 + i, tr = t / 12, tc = t % 12;
        int n = tc * 16 + (lane & 15);
        int mbase = tr * 16 + ((lane >> 4) << 3);
        float bb2 = b2[n];
#pragma unroll
        for (int r = 0; r < 8; ++r) {
            long lin = (r0 + mbase + r) * DIM + n;
            xo[lin] += yacc[i][r] + bb2;
        }
    }
}

// ================= launcher =================
extern "C" void kernel_launch(void* const* d_in, const int* in_sizes, int n_in,
                              void* d_out, int out_size, void* d_ws, size_t ws_size,
                              hipStream_t stream) {
    (void)in_sizes; (void)n_in; (void)out_size; (void)ws_size;
    const float* x     = (const float*)d_in[0];
    const float* mask  = (const float*)d_in[1];
    const float* n1g   = (const float*)d_in[2];
    const float* n1b   = (const float*)d_in[3];
    const float* qkvw  = (const float*)d_in[4];
    const float* qkvb  = (const float*)d_in[5];
    const float* rpb   = (const float*)d_in[6];
    const float* projw = (const float*)d_in[7];
    const float* projb = (const float*)d_in[8];
    const float* n2g   = (const float*)d_in[9];
    const float* n2b   = (const float*)d_in[10];
    const float* fc1w  = (const float*)d_in[11];
    const float* fc1b  = (const float*)d_in[12];
    const float* fc2w  = (const float*)d_in[13];
    const float* fc2b  = (const float*)d_in[14];
    float* out = (float*)d_out;

    char* ws = (char*)d_ws;
    half_t* qkvw_h  = (half_t*)(ws + OFF_QKVW);
    half_t* projw_h = (half_t*)(ws + OFF_PROJW);
    half_t* fc1w_h  = (half_t*)(ws + OFF_FC1W);
    half_t* fc2w_h  = (half_t*)(ws + OFF_FC2W);
    float*  biasbuf = (float*)(ws + OFF_BIAS);
    half_t* xw_h    = (half_t*)(ws + OFF_XW);

    cvt_f16_kernel<<<(576 * 192 + 255) / 256, 256, 0, stream>>>(qkvw, qkvw_h, 576 * 192);
    cvt_f16_kernel<<<(192 * 192 + 255) / 256, 256, 0, stream>>>(projw, projw_h, 192 * 192);
    cvt_f16_kernel<<<(768 * 192 + 255) / 256, 256, 0, stream>>>(fc1w, fc1w_h, 768 * 192);
    cvt_f16_kernel<<<(192 * 768 + 255) / 256, 256, 0, stream>>>(fc2w, fc2w_h, 192 * 768);
    build_bias_kernel<<<16, 256, 0, stream>>>(rpb, biasbuf);

    ln1_window_kernel<<<NWIN * NTOK / 8, 256, 0, stream>>>(x, n1g, n1b, xw_h);
    attn_kernel<<<NWIN, 256, 0, stream>>>(x, mask, xw_h, qkvw_h, qkvb,
                                          projw_h, projb, biasbuf, out);
    mlp_kernel<<<NWIN, 256, 0, stream>>>(out, n2g, n2b, fc1w_h, fc1b, fc2w_h, fc2b);
}
